// DenseTransformerDecoderLayer_48971217109654
// MI455X (gfx1250) — matrix operand order repaired and checked by reading, not compiled, not run on hardware
//
#include <hip/hip_runtime.h>

// ---------------- problem constants ----------------
#define T_     4096
#define HID_   2048
#define NH_    16
#define HD_    128
#define INTER_ 8192
#define NIN_   (3*HID_ + 2*INTER_)   // 22528 rows of W_in
#define NCMB_  (HID_ + INTER_)       // 10240 cols of combined
#define EPS_   1e-6f
#define SCALE_ 0.08838834764831845f  // 1/sqrt(128)

typedef __attribute__((ext_vector_type(16))) __bf16 v16bf;
typedef __attribute__((ext_vector_type(8)))  float  v8f;

union FragBF { unsigned int u[8]; v16bf v; };

#ifndef __has_builtin
#define __has_builtin(x) 0
#endif

// generic pointer (known to be in LDS) -> 32-bit LDS byte address
__device__ __forceinline__ unsigned int lds_addr32(const void* p) {
    return (unsigned int)(unsigned long long)(__attribute__((address_space(3))) const void*)p;
}

// 16-byte global -> LDS copy via CDNA5 async path (ASYNCcnt-tracked).
__device__ __forceinline__ void copy16_g2l(const unsigned short* g, unsigned short* l) {
    unsigned int loff = lds_addr32(l);
    asm volatile("global_load_async_to_lds_b128 %0, %1, off"
                 :: "v"(loff), "v"((unsigned long long)g)
                 : "memory");
}
__device__ __forceinline__ void async_wait() {
#if __has_builtin(__builtin_amdgcn_s_wait_asynccnt)
    __builtin_amdgcn_s_wait_asynccnt(0);
#else
    asm volatile("s_wait_asynccnt 0x0" ::: "memory");
#endif
}

__device__ __forceinline__ unsigned short f2bf(float f) {
    unsigned int u = __float_as_uint(f);
    unsigned int r = u + 0x7fffu + ((u >> 16) & 1u);   // round-to-nearest-even
    return (unsigned short)(r >> 16);
}
__device__ __forceinline__ float bf2f(unsigned short h) {
    return __uint_as_float(((unsigned int)h) << 16);
}

// ---------------- fp32 -> bf16 convert ----------------
__global__ __launch_bounds__(256)
void cvt_f32_bf16(const float* __restrict__ src, unsigned short* __restrict__ dst, long n) {
    long i = (long)blockIdx.x * 256 + threadIdx.x;
    if (i < n) dst[i] = f2bf(src[i]);
}

// ---------------- RMSNorm + age overrides -> bf16 h ----------------
__global__ __launch_bounds__(256)
void rmsnorm_kernel(const float* __restrict__ x, const float* __restrict__ ages,
                    const float* __restrict__ w, unsigned short* __restrict__ hB) {
    __shared__ float red[8];
    __shared__ float rsS;
    int row = blockIdx.x, tid = threadIdx.x;
    const float* xr = x + (long)row * HID_;
    float ss = 0.f;
    for (int i = tid; i < HID_; i += 256) { float v = xr[i]; ss += v * v; }
    #pragma unroll
    for (int off = 16; off > 0; off >>= 1) ss += __shfl_xor(ss, off, 32);
    if ((tid & 31) == 0) red[tid >> 5] = ss;
    __syncthreads();
    if (tid == 0) {
        float t = 0.f;
        #pragma unroll
        for (int i = 0; i < 8; ++i) t += red[i];
        rsS = rsqrtf(t / (float)HID_ + EPS_);
    }
    __syncthreads();
    float rs = rsS;
    float a = ages[row];
    for (int i = tid; i < HID_; i += 256) {
        float hv = xr[i] * rs * w[i];
        if (i == HID_ - 2) hv = a;
        if (i == HID_ - 1) hv = a * a;
        hB[(long)row * HID_ + i] = f2bf(hv);
    }
}

// ---------------- WMMA GEMM:  C[M,N] = A[M,K] * B[N,K]^T  (bf16 in, fp32 acc) ----
// block tile 256x128, BK=32, double-buffered async LDS staging.
// 8 waves as 4x2, each owning a 64x64 tile (16 accumulators).
template <bool BF16OUT>
__global__ __launch_bounds__(256)
void gemm_bf16_nt(const unsigned short* __restrict__ A,
                  const unsigned short* __restrict__ B,
                  void* __restrict__ Cout, int K, int ldc) {
    constexpr int BM = 256, BN = 128, BK = 32, LDA = BK + 2;
    __shared__ unsigned short sA[2][BM * LDA];   // 2 * 17408 B
    __shared__ unsigned short sB[2][BN * LDA];   // 2 *  8704 B

    const int tid  = threadIdx.x;
    const int lane = tid & 31;
    const int wid  = tid >> 5;
    const int waveM = wid >> 1;                  // 0..3
    const int waveN = wid & 1;                   // 0..1
    const int half  = lane >> 4;                 // 0/1
    const int l15   = lane & 15;
    const long blockRow = (long)blockIdx.y * BM;
    const long blockCol = (long)blockIdx.x * BN;

    v8f acc[4][4];
    #pragma unroll
    for (int mi = 0; mi < 4; ++mi)
        #pragma unroll
        for (int ni = 0; ni < 4; ++ni) acc[mi][ni] = (v8f)(0.f);

    auto stage = [&](int buf, int kt) {
        #pragma unroll
        for (int i = 0; i < 4; ++i) {            // A: 256x32 = 1024 uint4
            int c = tid + i * 256;
            int r = c >> 2, kc = (c & 3) * 8;
            copy16_g2l(A + (blockRow + r) * (long)K + kt + kc, &sA[buf][r * LDA + kc]);
        }
        #pragma unroll
        for (int i = 0; i < 2; ++i) {            // B: 128x32 = 512 uint4
            int c = tid + i * 256;
            int r = c >> 2, kc = (c & 3) * 8;
            copy16_g2l(B + (blockCol + r) * (long)K + kt + kc, &sB[buf][r * LDA + kc]);
        }
    };

    stage(0, 0);
    async_wait();
    __syncthreads();

    const int nk = K / BK;
    for (int it = 0; it < nk; ++it) {
        const int cur = it & 1, nxt = cur ^ 1;
        if (it + 1 < nk) stage(nxt, (it + 1) * BK);   // prefetch next tile (async)

        const unsigned short* cA = sA[cur];
        const unsigned short* cB = sB[cur];
        FragBF aF[4], bF[4];
        #pragma unroll
        for (int mi = 0; mi < 4; ++mi) {
            const unsigned short* base = cA + (waveM * 64 + mi * 16 + l15) * LDA;
            #pragma unroll
            for (int v = 0; v < 8; ++v) {
                int k = half * 16 + 2 * v;
                aF[mi].u[v] = *reinterpret_cast<const unsigned int*>(base + k);
            }
        }
        #pragma unroll
        for (int ni = 0; ni < 4; ++ni) {
            const unsigned short* base = cB + (waveN * 64 + ni * 16 + l15) * LDA;
            #pragma unroll
            for (int v = 0; v < 8; ++v) {
                int k = half * 16 + 2 * v;
                bF[ni].u[v] = *reinterpret_cast<const unsigned int*>(base + k);
            }
        }
        #pragma unroll
        for (int mi = 0; mi < 4; ++mi)
            #pragma unroll
            for (int ni = 0; ni < 4; ++ni)
                acc[mi][ni] = __builtin_amdgcn_wmma_f32_16x16x32_bf16(
                    false, aF[mi].v, false, bF[ni].v, (short)0, acc[mi][ni], false, false);

        async_wait();            // next-tile LDS writes landed
        __syncthreads();         // everyone done reading `cur`
    }

    // ---- epilogue (C layout: VGPR r -> row r (+8 for upper half), lane&15 -> col)
    #pragma unroll
    for (int mi = 0; mi < 4; ++mi)
        #pragma unroll
        for (int ni = 0; ni < 4; ++ni)
            #pragma unroll
            for (int r = 0; r < 8; ++r) {
                long row = blockRow + waveM * 64 + mi * 16 + half * 8 + r;
                long col = blockCol + waveN * 64 + ni * 16 + l15;
                float val = acc[mi][ni][r];
                if (BF16OUT) ((unsigned short*)Cout)[row * ldc + col] = f2bf(val);
                else         ((float*)Cout)[row * ldc + col] = val;
            }
}

// ---------------- FF: silu(g1)*g2 -> combined[:, 2048:] ----------------
__global__ __launch_bounds__(256)
void ff_kernel(const unsigned short* __restrict__ tB, unsigned short* __restrict__ comb) {
    long i = (long)blockIdx.x * 256 + threadIdx.x;
    if (i >= (long)T_ * INTER_) return;
    long row = i / INTER_, c = i % INTER_;
    float g1 = bf2f(tB[row * NIN_ + c]);
    float g2 = bf2f(tB[row * NIN_ + INTER_ + c]);
    float s = g1 / (1.f + __expf(-g1));
    comb[row * NCMB_ + HID_ + c] = f2bf(s * g2);
}

// ---------------- RoPE + QKV scatter into [NH][T][HD] ----------------
__global__ __launch_bounds__(256)
void rope_qkv_kernel(const unsigned short* __restrict__ tB,
                     const float* __restrict__ sinT, const float* __restrict__ cosT,
                     unsigned short* __restrict__ qR, unsigned short* __restrict__ kR,
                     unsigned short* __restrict__ vR) {
    long i = (long)blockIdx.x * 256 + threadIdx.x;
    if (i >= (long)T_ * NH_ * (HD_ / 2)) return;
    int d2  = (int)(i % (HD_ / 2));
    int h   = (int)((i / (HD_ / 2)) % NH_);
    long row = i / ((HD_ / 2) * NH_);
    int d0 = 2 * d2;
    long base = row * NIN_ + 2 * INTER_ + h * HD_ + d0;
    float q0 = bf2f(tB[base]),              q1 = bf2f(tB[base + 1]);
    float k0 = bf2f(tB[base + HID_]),       k1 = bf2f(tB[base + HID_ + 1]);
    float v0 = bf2f(tB[base + 2 * HID_]),   v1 = bf2f(tB[base + 2 * HID_ + 1]);
    float c0 = cosT[row * HD_ + d0], c1 = cosT[row * HD_ + d0 + 1];
    float s0 = sinT[row * HD_ + d0], s1 = sinT[row * HD_ + d0 + 1];
    float qe = q0 * c0 - q1 * s0, qo = q1 * c1 + q0 * s1;
    float ke = k0 * c0 - k1 * s0, ko = k1 * c1 + k0 * s1;
    long o = ((long)h * T_ + row) * HD_ + d0;
    qR[o] = f2bf(qe * SCALE_); qR[o + 1] = f2bf(qo * SCALE_);   // fold score scale into q
    kR[o] = f2bf(ke);          kR[o + 1] = f2bf(ko);
    vR[o] = f2bf(v0);          vR[o + 1] = f2bf(v1);
}

// ---------------- FlashAttention (causal) -> combined[:, 0:2048] ----------------
// block = (head, 128-query tile); wave w owns 16 queries. K/V tiles of 32 keys in LDS.
__global__ __launch_bounds__(256)
void attn_kernel(const unsigned short* __restrict__ qR, const unsigned short* __restrict__ kR,
                 const unsigned short* __restrict__ vR, unsigned short* __restrict__ comb) {
    constexpr int LDK = HD_ + 2;     // sK: [key 0..31][hd 0..127]
    constexpr int LDV = 34;          // sV: [hd 0..127][key 0..31] (transposed)
    constexpr int LDP = 34;          // per-wave P patch 16x32
    __shared__ unsigned short sK[32 * LDK];
    __shared__ unsigned short sV[HD_ * LDV];
    __shared__ unsigned short sP[8][16 * LDP];

    const int tid = threadIdx.x, lane = tid & 31, wid = tid >> 5;
    const int half = lane >> 4, l15 = lane & 15;
    const int head = blockIdx.y;
    const int qBase = blockIdx.x * 128;
    const int waveQ = qBase + wid * 16;
    const unsigned short* qH = qR + (long)head * T_ * HD_;
    const unsigned short* kH = kR + (long)head * T_ * HD_;
    const unsigned short* vH = vR + (long)head * T_ * HD_;

    // Q as 4 A-fragments (16 x 32 over HD)
    FragBF qF[4];
    #pragma unroll
    for (int f = 0; f < 4; ++f) {
        const unsigned short* rp = qH + (long)(waveQ + l15) * HD_ + f * 32 + half * 16;
        uint4 lo = *reinterpret_cast<const uint4*>(rp);
        uint4 hi = *reinterpret_cast<const uint4*>(rp + 8);
        qF[f].u[0] = lo.x; qF[f].u[1] = lo.y; qF[f].u[2] = lo.z; qF[f].u[3] = lo.w;
        qF[f].u[4] = hi.x; qF[f].u[5] = hi.y; qF[f].u[6] = hi.z; qF[f].u[7] = hi.w;
    }

    v8f O[8];
    #pragma unroll
    for (int nt = 0; nt < 8; ++nt) O[nt] = (v8f)(0.f);
    float mrow[8], lrow[8];
    #pragma unroll
    for (int r = 0; r < 8; ++r) { mrow[r] = -1e30f; lrow[r] = 0.f; }

    const int jtN = blockIdx.x * 4 + 4;        // key tiles covering keys 0..qBase+127
    const int myKeyMax = waveQ + 15;

    for (int jt = 0; jt < jtN; ++jt) {
        const int keyBase = jt * 32;
        // ---- stage K (async row-major) and V (manual transpose) tiles
        #pragma unroll
        for (int i = 0; i < 2; ++i) {
            int c = tid + i * 256;             // 0..511
            int r = c >> 4, kc = (c & 15) * 8;
            copy16_g2l(kH + (long)(keyBase + r) * HD_ + kc, sK + r * LDK + kc);
            uint4 dv = *reinterpret_cast<const uint4*>(vH + (long)(keyBase + r) * HD_ + kc);
            unsigned short e[8];
            __builtin_memcpy(e, &dv, 16);
            #pragma unroll
            for (int j = 0; j < 8; ++j) sV[(kc + j) * LDV + r] = e[j];
        }
        async_wait();
        __syncthreads();

        if (keyBase <= myKeyMax) {             // wave-uniform causal skip
            // ---- S = q * k^T  (2 key sub-tiles of 16, contraction HD in 4 steps)
            v8f S[2]; S[0] = (v8f)(0.f); S[1] = (v8f)(0.f);
            #pragma unroll
            for (int f = 0; f < 4; ++f) {
                #pragma unroll
                for (int nt = 0; nt < 2; ++nt) {
                    FragBF bK;
                    const unsigned short* base = sK + (nt * 16 + l15) * LDK + f * 32 + half * 16;
                    #pragma unroll
                    for (int v = 0; v < 8; ++v)
                        bK.u[v] = *reinterpret_cast<const unsigned int*>(base + 2 * v);
                    S[nt] = __builtin_amdgcn_wmma_f32_16x16x32_bf16(
                        false, qF[f].v, false, bK.v, (short)0, S[nt], false, false);
                }
            }
            // ---- online softmax (rows live in half-16 lane groups)
            float p0a[8], p1a[8];
            #pragma unroll
            for (int r = 0; r < 8; ++r) {
                int qrow = waveQ + half * 8 + r;
                float s0 = (keyBase + l15      <= qrow) ? S[0][r] : -1e30f;
                float s1 = (keyBase + 16 + l15 <= qrow) ? S[1][r] : -1e30f;
                float m8 = fmaxf(s0, s1);
                #pragma unroll
                for (int off = 1; off < 16; off <<= 1) m8 = fmaxf(m8, __shfl_xor(m8, off, 32));
                float mnew = fmaxf(mrow[r], m8);
                float alpha = __expf(mrow[r] - mnew);
                float p0 = __expf(s0 - mnew), p1 = __expf(s1 - mnew);
                float rs = p0 + p1;
                #pragma unroll
                for (int off = 1; off < 16; off <<= 1) rs += __shfl_xor(rs, off, 32);
                lrow[r] = lrow[r] * alpha + rs;
                mrow[r] = mnew;
                #pragma unroll
                for (int nt = 0; nt < 8; ++nt) O[nt][r] *= alpha;
                p0a[r] = p0; p1a[r] = p1;
            }
            // ---- P: C-layout -> A-layout via this wave's LDS patch
            unsigned short* myP = sP[wid];
            #pragma unroll
            for (int r = 0; r < 8; ++r) {
                myP[(half * 8 + r) * LDP + l15]      = f2bf(p0a[r]);
                myP[(half * 8 + r) * LDP + 16 + l15] = f2bf(p1a[r]);
            }
            FragBF pF;
            #pragma unroll
            for (int v = 0; v < 8; ++v) {
                int k = half * 16 + 2 * v;
                pF.u[v] = *reinterpret_cast<const unsigned int*>(myP + l15 * LDP + k);
            }
            // ---- O += P * V  (contraction = 32 keys, 8 HD sub-tiles)
            #pragma unroll
            for (int nt = 0; nt < 8; ++nt) {
                FragBF bV;
                const unsigned short* base = sV + (nt * 16 + l15) * LDV + half * 16;
                #pragma unroll
                for (int v = 0; v < 8; ++v)
                    bV.u[v] = *reinterpret_cast<const unsigned int*>(base + 2 * v);
                O[nt] = __builtin_amdgcn_wmma_f32_16x16x32_bf16(
                    false, pF.v, false, bV.v, (short)0, O[nt], false, false);
            }
        }
        __syncthreads();
    }

    // ---- normalize + store to combined[:, head*128 ...]
    #pragma unroll
    for (int r = 0; r < 8; ++r) {
        float inv = 1.f / lrow[r];
        long row = waveQ + half * 8 + r;
        #pragma unroll
        for (int nt = 0; nt < 8; ++nt)
            comb[row * NCMB_ + head * HD_ + nt * 16 + l15] = f2bf(O[nt][r] * inv);
    }
}

// ---------------- launcher ----------------
extern "C" void kernel_launch(void* const* d_in, const int* in_sizes, int n_in,
                              void* d_out, int out_size, void* d_ws, size_t ws_size,
                              hipStream_t stream) {
    (void)in_sizes; (void)n_in; (void)out_size; (void)ws_size;
    const float* x      = (const float*)d_in[0];
    const float* ages   = (const float*)d_in[1];
    const float* sinT   = (const float*)d_in[2];
    const float* cosT   = (const float*)d_in[3];
    const float* norm_w = (const float*)d_in[4];
    const float* W_in   = (const float*)d_in[5];
    const float* W_out  = (const float*)d_in[6];

    unsigned char* ws = (unsigned char*)d_ws;
    size_t off = 0;
    auto take = [&](size_t bytes) -> void* {
        void* p = ws + off;
        off += (bytes + 255) & ~(size_t)255;
        return p;
    };
    unsigned short* WinB  = (unsigned short*)take((size_t)NIN_ * HID_ * 2);
    unsigned short* WoutB = (unsigned short*)take((size_t)HID_ * NCMB_ * 2);
    unsigned short* hB    = (unsigned short*)take((size_t)T_ * HID_ * 2);
    unsigned short* tB    = (unsigned short*)take((size_t)T_ * NIN_ * 2);
    unsigned short* qR    = (unsigned short*)take((size_t)NH_ * T_ * HD_ * 2);
    unsigned short* kRp   = (unsigned short*)take((size_t)NH_ * T_ * HD_ * 2);
    unsigned short* vRp   = (unsigned short*)take((size_t)NH_ * T_ * HD_ * 2);
    unsigned short* comb  = (unsigned short*)take((size_t)T_ * NCMB_ * 2);

    { long n = (long)NIN_ * HID_;
      cvt_f32_bf16<<<(unsigned)((n + 255) / 256), 256, 0, stream>>>(W_in, WinB, n); }
    { long n = (long)HID_ * NCMB_;
      cvt_f32_bf16<<<(unsigned)((n + 255) / 256), 256, 0, stream>>>(W_out, WoutB, n); }

    rmsnorm_kernel<<<T_, 256, 0, stream>>>(x, ages, norm_w, hB);

    gemm_bf16_nt<true><<<dim3(NIN_ / 128, T_ / 256), 256, 0, stream>>>(
        hB, WinB, tB, HID_, NIN_);

    { long n = (long)T_ * INTER_;
      ff_kernel<<<(unsigned)((n + 255) / 256), 256, 0, stream>>>(tB, comb); }
    { long n = (long)T_ * NH_ * (HD_ / 2);
      rope_qkv_kernel<<<(unsigned)((n + 255) / 256), 256, 0, stream>>>(
          tB, sinT, cosT, qR, kRp, vRp); }

    attn_kernel<<<dim3(T_ / 128, NH_), 256, 0, stream>>>(qR, kRp, vRp, comb);

    gemm_bf16_nt<false><<<dim3(HID_ / 128, T_ / 256), 256, 0, stream>>>(
        comb, WoutB, d_out, NCMB_, HID_);
}
